// GLORYServer_25494925869146
// MI455X (gfx1250) — matrix operand order repaired and checked by reading, not compiled, not run on hardware
//
#include <hip/hip_runtime.h>
#include <hip/hip_bf16.h>

// ---------------- problem constants ----------------
#define NNODES  50000
#define NEDGES  200000
#define DIM     400          // D
#define DIM3    1200         // 3D
#define KP      416          // D padded to multiple of 32 (13 k-tiles)
#define KT      13           // KP / 32
#define NBD     416          // D padded to multiple of 32 for B cols
#define NB3     1216         // 3D padded to multiple of 32 for B cols
#define MT32    1563         // ceil(NNODES / 32)

typedef __attribute__((ext_vector_type(16))) __bf16 v16bf;
typedef __attribute__((ext_vector_type(8)))  float  v8f;

// ---- cast fp32 -> bf16 directly into WMMA A-fragment layout ----
// Ap[((row*KT + kt)*32) + g*16 + e]  holds  A[row][K(kt,g,e)] where
//   e in 0..7  -> K = kt*32 + 8g + e
//   e in 8..15 -> K = kt*32 + 16 + 8g + (e-8)
// so lane (m + 16g) reads its whole 16-element fragment as ONE 32B load.
__global__ __launch_bounds__(256) void cast_pack_a_bf16(const float* __restrict__ in,
                                                        __bf16* __restrict__ outp) {
  int idx = blockIdx.x * blockDim.x + threadIdx.x;           // over NNODES*KP
  if (idx >= NNODES * KP) return;
  int t   = idx & 31;                  // slot within fragment pair
  int kt  = (idx >> 5) % KT;
  int row = (idx >> 5) / KT;
  int g   = t >> 4;
  int e   = t & 15;
  int k   = kt * 32 + ((e < 8) ? (8 * g + e) : (16 + 8 * g + (e - 8)));
  float v = (k < DIM) ? in[(size_t)row * DIM + k] : 0.0f;
  outp[idx] = (__bf16)v;
}

// ---------------- pack weights into B-fragment layout [kt][col][32] ----------
// transpose==0: B[k][n] = in[k*Ncols + n]   (weight[l], [D][D] row-major)
// transpose==1: B[k][n] = in[n*DIM + k]     (w_ih / w_hh are [3D][D] row-major)
__global__ __launch_bounds__(256) void pack_b_bf16(const float* __restrict__ in,
                                                   __bf16* __restrict__ outp,
                                                   int Ncols, int NB, int transpose) {
  int idx = blockIdx.x * blockDim.x + threadIdx.x;           // over KT*NB*32
  if (idx >= KT * NB * 32) return;
  int kk  = idx & 31;
  int col = (idx >> 5) % NB;
  int kt  = (idx >> 5) / NB;
  int k   = kt * 32 + (((kk & 15) < 8) ? (8 * (kk >> 4) + (kk & 15)) : (16 + 8 * (kk >> 4) + ((kk & 15) - 8)));
  float v = 0.0f;
  if (k < DIM && col < Ncols)
    v = transpose ? in[(size_t)col * DIM + k] : in[(size_t)k * Ncols + col];
  outp[idx] = (__bf16)v;
}

// ---------------- BF16 WMMA GEMM: C[M x Ncols] = Ap * Bp ---------------------
// Each wave computes a 32(M) x 32(N) C tile: 2 A-frags x 2 B-frags -> 4 WMMAs
// per k-tile at 1 VMEM load per WMMA. Block = 8 waves covering 256 cols.
__global__ __launch_bounds__(256) void gemm_bf16_wmma(const __bf16* __restrict__ Ap,
                                                      const __bf16* __restrict__ Bp,
                                                      float* __restrict__ C,
                                                      int Ncols, int NB) {
  const int lane  = threadIdx.x & 31;
  const int wave  = threadIdx.x >> 5;
  const int m16   = lane & 15;    // A: row in tile; B/C: col in tile
  const int g     = lane >> 4;    // half-wave selector
  const int ncol0 = blockIdx.x * 256 + wave * 32;
  if (ncol0 >= NB) return;
  const int mbase = blockIdx.y * 32;
  // clamp A-load rows for the (partial) last M-block; stores are guarded below
  const int rA0 = min(mbase + m16,      NNODES - 1);
  const int rA1 = min(mbase + 16 + m16, NNODES - 1);
  const __bf16* Afrag0 = Ap + (size_t)rA0 * KP + 16 * g;   // + kt*32 per tile
  const __bf16* Afrag1 = Ap + (size_t)rA1 * KP + 16 * g;
  const __bf16* Bfrag0 = Bp + (size_t)(ncol0 + m16)      * 32 + 16 * g;
  const __bf16* Bfrag1 = Bp + (size_t)(ncol0 + 16 + m16) * 32 + 16 * g;

  v8f c00 = {}; v8f c01 = {}; v8f c10 = {}; v8f c11 = {};
  for (int kt = 0; kt < KT; ++kt) {
    v16bf a0 = *(const v16bf*)(Afrag0 + kt * 32);
    v16bf a1 = *(const v16bf*)(Afrag1 + kt * 32);
    v16bf b0 = *(const v16bf*)(Bfrag0 + (size_t)kt * NB * 32);
    v16bf b1 = *(const v16bf*)(Bfrag1 + (size_t)kt * NB * 32);
    c00 = __builtin_amdgcn_wmma_f32_16x16x32_bf16(false, a0, false, b0, (short)0, c00, false, false);
    c01 = __builtin_amdgcn_wmma_f32_16x16x32_bf16(false, a0, false, b1, (short)0, c01, false, false);
    c10 = __builtin_amdgcn_wmma_f32_16x16x32_bf16(false, a1, false, b0, (short)0, c10, false, false);
    c11 = __builtin_amdgcn_wmma_f32_16x16x32_bf16(false, a1, false, b1, (short)0, c11, false, false);
  }
  // C/D layout: element r of lane (n + 16g) -> row 8g + r, col n
  const int orow0 = mbase + 8 * g;        // rows for c00/c01
  const int orow1 = mbase + 16 + 8 * g;   // rows for c10/c11
  const int col0  = ncol0 + m16;
  const int col1  = col0 + 16;
#pragma unroll
  for (int r = 0; r < 8; ++r) {
    if (orow0 + r < NNODES) {
      size_t base = (size_t)(orow0 + r) * Ncols;
      if (col0 < Ncols) C[base + col0] = c00[r];
      if (col1 < Ncols) C[base + col1] = c01[r];
    }
    if (orow1 + r < NNODES) {
      size_t base = (size_t)(orow1 + r) * Ncols;
      if (col0 < Ncols) C[base + col0] = c10[r];
      if (col1 < Ncols) C[base + col1] = c11[r];
    }
  }
}

// ---------------- zero buffer ----------------
__global__ __launch_bounds__(256) void zero_f32(float* __restrict__ p, int n) {
  int idx = blockIdx.x * blockDim.x + threadIdx.x;
  if (idx < n) p[idx] = 0.0f;
}

// ---------------- edge scatter-add: agg[dst] += m[src] ----------------
__global__ __launch_bounds__(256) void scatter_add(const float* __restrict__ m,
                                                   const int* __restrict__ eidx,
                                                   float* __restrict__ agg) {
  int t = blockIdx.x * blockDim.x + threadIdx.x;             // over NEDGES * 100
  if (t >= NEDGES * 100) return;
  int e = t / 100;
  int c = (t % 100) * 4;
  int s = eidx[e];                 // src row
  int d = eidx[NEDGES + e];        // dst row
  float4 v = *(const float4*)(m + (size_t)s * DIM + c);
  float* p = agg + (size_t)d * DIM + c;
  atomicAdd(p + 0, v.x);
  atomicAdd(p + 1, v.y);
  atomicAdd(p + 2, v.z);
  atomicAdd(p + 3, v.w);
}

// ---------------- fused GRU gates ----------------
__global__ __launch_bounds__(256) void gru_gates(const float* __restrict__ gi,
                                                 const float* __restrict__ gh,
                                                 const float* __restrict__ b_ih,
                                                 const float* __restrict__ b_hh,
                                                 const float* __restrict__ hprev,
                                                 float* __restrict__ hout) {
  int idx = blockIdx.x * blockDim.x + threadIdx.x;           // over NNODES*DIM
  if (idx >= NNODES * DIM) return;
  int j   = idx % DIM;
  size_t rb = (size_t)(idx / DIM) * DIM3;
  float ir = gi[rb + j]            + b_ih[j];
  float iz = gi[rb + DIM + j]      + b_ih[DIM + j];
  float in = gi[rb + 2 * DIM + j]  + b_ih[2 * DIM + j];
  float hr = gh[rb + j]            + b_hh[j];
  float hz = gh[rb + DIM + j]      + b_hh[DIM + j];
  float hn = gh[rb + 2 * DIM + j]  + b_hh[2 * DIM + j];
  float r = 1.0f / (1.0f + __expf(-(ir + hr)));
  float z = 1.0f / (1.0f + __expf(-(iz + hz)));
  float n = tanhf(in + r * hn);
  hout[idx] = (1.0f - z) * n + z * hprev[idx];
}

// ---------------- host orchestration ----------------
extern "C" void kernel_launch(void* const* d_in, const int* in_sizes, int n_in,
                              void* d_out, int out_size, void* d_ws, size_t ws_size,
                              hipStream_t stream) {
  const float* x      = (const float*)d_in[0];   // [N, D]
  const float* weight = (const float*)d_in[1];   // [3, D, D]
  const float* w_ih   = (const float*)d_in[2];   // [3D, D]
  const float* w_hh   = (const float*)d_in[3];   // [3D, D]
  const float* b_ih   = (const float*)d_in[4];   // [3D]
  const float* b_hh   = (const float*)d_in[5];   // [3D]
  const int*   eidx   = (const int*)d_in[6];     // [2, E]
  (void)in_sizes; (void)n_in; (void)out_size; (void)ws_size;
  float* out = (float*)d_out;

  // ---- carve workspace (256B aligned slices) ----
  char* ws = (char*)d_ws;
  size_t off = 0;
  auto carve = [&](size_t bytes) -> char* {
    char* p = ws + off;
    off += (bytes + 255) & ~(size_t)255;
    return p;
  };
  __bf16* hb   = (__bf16*)carve((size_t)NNODES * KP * 2);
  __bf16* aggb = (__bf16*)carve((size_t)NNODES * KP * 2);
  float*  mbuf = (float*) carve((size_t)NNODES * DIM * 4);
  float*  agg  = (float*) carve((size_t)NNODES * DIM * 4);
  float*  gh   = (float*) carve((size_t)NNODES * DIM3 * 4);
  float*  gi   = (float*) carve((size_t)NNODES * DIM3 * 4);
  float*  h1   = (float*) carve((size_t)NNODES * DIM * 4);
  float*  h2   = (float*) carve((size_t)NNODES * DIM * 4);
  __bf16* Wp   = (__bf16*)carve((size_t)3 * KT * NBD * 32 * 2);
  __bf16* Wihp = (__bf16*)carve((size_t)KT * NB3 * 32 * 2);
  __bf16* Whhp = (__bf16*)carve((size_t)KT * NB3 * 32 * 2);

  const int packD_elems = KT * NBD * 32;
  const int pack3_elems = KT * NB3 * 32;

  // ---- pre-pack all weight matrices to B-fragment bf16 layout ----
  for (int l = 0; l < 3; ++l)
    pack_b_bf16<<<(packD_elems + 255) / 256, 256, 0, stream>>>(
        weight + (size_t)l * DIM * DIM, Wp + (size_t)l * packD_elems, DIM, NBD, 0);
  pack_b_bf16<<<(pack3_elems + 255) / 256, 256, 0, stream>>>(w_ih, Wihp, DIM3, NB3, 1);
  pack_b_bf16<<<(pack3_elems + 255) / 256, 256, 0, stream>>>(w_hh, Whhp, DIM3, NB3, 1);

  const int castN  = NNODES * KP;
  const int gateN  = NNODES * DIM;
  const int scatN  = NEDGES * 100;

  const float* hin = x;
  float* houts[3] = { h1, h2, out };

  for (int l = 0; l < 3; ++l) {
    // 1) h -> bf16, packed to A-fragment layout
    cast_pack_a_bf16<<<(castN + 255) / 256, 256, 0, stream>>>(hin, hb);
    // 2) m = h @ W[l]
    gemm_bf16_wmma<<<dim3((NBD + 255) / 256, MT32), 256, 0, stream>>>(
        hb, Wp + (size_t)l * packD_elems, mbuf, DIM, NBD);
    // 3) agg = segment_sum(m[src], dst)
    zero_f32<<<(gateN + 255) / 256, 256, 0, stream>>>(agg, gateN);
    scatter_add<<<(scatN + 255) / 256, 256, 0, stream>>>(mbuf, eidx, agg);
    // 4) agg -> bf16, packed to A-fragment layout
    cast_pack_a_bf16<<<(castN + 255) / 256, 256, 0, stream>>>(agg, aggb);
    // 5) gh = h @ w_hh^T
    gemm_bf16_wmma<<<dim3((NB3 + 255) / 256, MT32), 256, 0, stream>>>(
        hb, Whhp, gh, DIM3, NB3);
    // 6) gi = agg @ w_ih^T
    gemm_bf16_wmma<<<dim3((NB3 + 255) / 256, MT32), 256, 0, stream>>>(
        aggb, Wihp, gi, DIM3, NB3);
    // 7) fused GRU gates -> next h
    gru_gates<<<(gateN + 255) / 256, 256, 0, stream>>>(gi, gh, b_ih, b_hh, hin, houts[l]);
    hin = houts[l];
  }
}
